// LSR_29618094473315
// MI455X (gfx1250) — compile-verified, matrix operand order checked by reading, not run on hardware
//
#include <hip/hip_runtime.h>
#include <stdint.h>

// ---------------------------------------------------------------------------
// CDNA5 (gfx1250, wave32) implementation of the LSR forward pass.
// All dense GEMMs go through v_wmma_f32_16x16x32_bf16 (f32 accum).
// ---------------------------------------------------------------------------

typedef unsigned short u16;
typedef __attribute__((ext_vector_type(16))) __bf16 v16bf;
typedef __attribute__((ext_vector_type(8)))  float  v8f;

// ---------------- device helpers ----------------

__device__ __forceinline__ float wred_sum(float v) {
#pragma unroll
  for (int o = 16; o > 0; o >>= 1) v += __shfl_xor(v, o, 32);
  return v;
}
__device__ __forceinline__ float wred_max(float v) {
#pragma unroll
  for (int o = 16; o > 0; o >>= 1) v = fmaxf(v, __shfl_xor(v, o, 32));
  return v;
}
__device__ __forceinline__ u16 f2bf(float f) {
  unsigned u = __float_as_uint(f);
  u = (u + 0x7FFFu + ((u >> 16) & 1u)) >> 16;   // round-to-nearest-even
  return (u16)u;
}

// ---------------- WMMA GEMM ----------------
// C[z] = epilogue( alpha * A[z] @ B[z] )
//   A: bf16 [M,K] row-major (lda), B: bf16 [K,N] row-major (ldb) or, if
//   transB, [N,K] row-major.  Batch offset per z: (z/innerN)*sXo+(z%innerN)*sXi.
//   epilogue: v = acc*alpha + bias[col]*bscale; if rowdiv: v /= rowdiv[row];
//             if relu: v = max(v,0).

#define BM 128
#define BN 128
#define BKK 32
#define LDSK 40   // padded K-stride in LDS (bank-conflict free b128 reads)

union Frag { v16bf v; uint4 q[2]; };

__global__ __launch_bounds__(256) void wmma_gemm_kernel(
    const u16* __restrict__ A, const u16* __restrict__ B, float* __restrict__ C,
    const float* __restrict__ bias, const float* __restrict__ rowdiv,
    int M, int N, int K, int lda, int ldb, int ldc,
    long sAo, long sAi, int innerN,
    long sBo, long sBi, long sCo, long sCi,
    float alpha, float bscale, int transB, int relu)
{
  const int z  = blockIdx.z;
  const int zo = z / innerN, zi = z % innerN;
  const u16* Ab = A + zo * sAo + zi * sAi;
  const u16* Bb = B + zo * sBo + zi * sBi;
  float*     Cb = C + zo * sCo + zi * sCi;

  const int tileM = blockIdx.y * BM;
  const int tileN = blockIdx.x * BN;
  const int tid  = threadIdx.x;
  const int lane = tid & 31;
  const int wave = tid >> 5;      // 8 waves: 4 along M x 2 along N
  const int wM = wave & 3;        // 32 rows each
  const int wN = wave >> 2;       // 64 cols each

  __shared__ u16 lA[BM * LDSK];
  __shared__ u16 lB[BN * LDSK];

  v8f acc[2][4];
  const v8f vzero = {0.f, 0.f, 0.f, 0.f, 0.f, 0.f, 0.f, 0.f};
#pragma unroll
  for (int i = 0; i < 2; ++i)
#pragma unroll
    for (int j = 0; j < 4; ++j) acc[i][j] = vzero;

  const int nK = K / BKK;
  for (int kt = 0; kt < nK; ++kt) {
    const int k0 = kt * BKK;

    // ---- A tile: [BM][BKK] bf16, direct row-major copy (b128 chunks) ----
#pragma unroll
    for (int c = tid; c < 512; c += 256) {
      int r = c >> 2, ch = c & 3;
      int gr = tileM + r, gc = k0 + ch * 8;
      uint4 val = make_uint4(0u, 0u, 0u, 0u);
      if (gr < M) val = *(const uint4*)(Ab + (long)gr * lda + gc);
      *(uint4*)(&lA[r * LDSK + ch * 8]) = val;
    }
    // ---- B tile into LDS as [N][K] ----
    if (transB) {
#pragma unroll
      for (int c = tid; c < 512; c += 256) {
        int r = c >> 2, ch = c & 3;          // r = n index
        int gn = tileN + r, gk = k0 + ch * 8;
        uint4 val = make_uint4(0u, 0u, 0u, 0u);
        if (gn < N) val = *(const uint4*)(Bb + (long)gn * ldb + gk);
        *(uint4*)(&lB[r * LDSK + ch * 8]) = val;
      }
    } else {
#pragma unroll
      for (int c = tid; c < 512; c += 256) {
        int kk = c >> 4, ch = c & 15;        // 32 k-rows x 16 chunks of 8 n
        int gk = k0 + kk, gn0 = tileN + ch * 8;
        u16 tmp[8];
        if (gn0 + 7 < N) {
          *(uint4*)tmp = *(const uint4*)(Bb + (long)gk * ldb + gn0);
        } else {
#pragma unroll
          for (int j = 0; j < 8; ++j)
            tmp[j] = (gn0 + j < N) ? Bb[(long)gk * ldb + gn0 + j] : (u16)0;
        }
#pragma unroll
        for (int j = 0; j < 8; ++j) lB[(ch * 8 + j) * LDSK + kk] = tmp[j];
      }
    }
    // hint the next K-tile of A into cache
    if (kt + 1 < nK) {
      int gr = tileM + (tid >> 1);
      if (gr < M) __builtin_prefetch(Ab + (long)gr * lda + k0 + BKK, 0, 3);
    }
    __syncthreads();

    // ---- fragments + WMMA ----
    // 16-bit A/B layout: lane row = lane&15; K-half base = (lane&16)>>1;
    // v16 halves 0..7 -> K[base..base+7], 8..15 -> K[base+16..base+23].
    const int kb   = (lane & 16) >> 1;
    const int mrow = lane & 15;
    Frag aF[2], bF[4];
#pragma unroll
    for (int i = 0; i < 2; ++i) {
      int r = wM * 32 + i * 16 + mrow;
      aF[i].q[0] = *(const uint4*)(&lA[r * LDSK + kb]);
      aF[i].q[1] = *(const uint4*)(&lA[r * LDSK + kb + 16]);
    }
#pragma unroll
    for (int j = 0; j < 4; ++j) {
      int n = wN * 64 + j * 16 + mrow;
      bF[j].q[0] = *(const uint4*)(&lB[n * LDSK + kb]);
      bF[j].q[1] = *(const uint4*)(&lB[n * LDSK + kb + 16]);
    }
#pragma unroll
    for (int i = 0; i < 2; ++i)
#pragma unroll
      for (int j = 0; j < 4; ++j)
        acc[i][j] = __builtin_amdgcn_wmma_f32_16x16x32_bf16(
            false, aF[i].v, false, bF[j].v, (short)0, acc[i][j], false, false);
    __syncthreads();
  }

  // ---- epilogue ----
  const int rofs = (lane & 16) ? 8 : 0;
#pragma unroll
  for (int i = 0; i < 2; ++i)
#pragma unroll
    for (int j = 0; j < 4; ++j) {
      int baseM = tileM + wM * 32 + i * 16 + rofs;
      int col   = tileN + wN * 64 + j * 16 + (lane & 15);
      if (col >= N) continue;
      float bv = bias ? bias[col] * bscale : 0.0f;
#pragma unroll
      for (int r = 0; r < 8; ++r) {
        int row = baseM + r;
        if (row >= M) continue;
        float v = acc[i][j][r] * alpha + bv;
        if (rowdiv) v /= rowdiv[row];
        if (relu) v = fmaxf(v, 0.0f);
        Cb[(long)row * ldc + col] = v;
      }
    }
}

// ---------------- generic elementwise: s = a (+ b); df=s; dh=bf16(sum or a) --

__global__ void ew_kernel(const float* __restrict__ a, int lda,
                          const float* __restrict__ b, int ldb,
                          float* __restrict__ df, int lddf,
                          u16* __restrict__ dh, int lddh,
                          long rows, long cols, int dhFromSum)
{
  long total = rows * cols;
  for (long idx = (long)blockIdx.x * blockDim.x + threadIdx.x; idx < total;
       idx += (long)gridDim.x * blockDim.x) {
    long r = idx / cols, c = idx % cols;
    float av = a[r * lda + c];
    float s  = av + (b ? b[r * ldb + c] : 0.0f);
    if (df) df[r * lddf + c] = s;
    if (dh) dh[r * lddh + c] = f2bf(dhFromSum ? s : av);
  }
}

// ---------------- root scores: f[(b*2+t)*512+l] = x[b,l,:]·W_t + b_t --------

__global__ void root_kernel(const float* __restrict__ x,
                            const float* __restrict__ W0, const float* __restrict__ b0,
                            const float* __restrict__ W1, const float* __restrict__ b1,
                            float* __restrict__ f)
{
  int gw = (blockIdx.x * blockDim.x + threadIdx.x) >> 5;
  int lane = threadIdx.x & 31;
  if (gw >= 16384) return;
  int bt = gw >> 9, l = gw & 511;
  int b = bt >> 1, t = bt & 1;
  const float* W  = t ? W1 : W0;
  const float* xr = x + ((long)b * 512 + l) * 768;
  float s = 0.f;
#pragma unroll
  for (int c = 0; c < 24; ++c) { int j = c * 32 + lane; s += xr[j] * W[j]; }
  s = wred_sum(s);
  if (lane == 0) f[gw] = s + (t ? b1[0] : b0[0]);
}

// -------- masked softmax over last dim, then A = exp(p) * (1 - eye) ---------

__global__ void softmax_expA_kernel(float* __restrict__ S, const int* __restrict__ mask)
{
  int gw = (blockIdx.x * blockDim.x + threadIdx.x) >> 5;
  int lane = threadIdx.x & 31;
  if (gw >= 16384) return;                 // B*T*L rows
  int bt = gw >> 9, i = gw & 511, b = bt >> 1;
  float* row = S + (long)gw * 512;
  const int* mrow = mask + b * 512;
  float v[16]; float mx = -3.0e38f;
#pragma unroll
  for (int c = 0; c < 16; ++c) {
    int j = c * 32 + lane;
    float s = (mrow[j] == 0) ? -1.0e9f : row[j];
    v[c] = s; mx = fmaxf(mx, s);
  }
  mx = wred_max(mx);
  float sum = 0.f;
#pragma unroll
  for (int c = 0; c < 16; ++c) { v[c] = __expf(v[c] - mx); sum += v[c]; }
  sum = wred_sum(sum);
  float inv = 1.0f / sum;
#pragma unroll
  for (int c = 0; c < 16; ++c) {
    int j = c * 32 + lane;
    float p = v[c] * inv;
    row[j] = (j == i) ? 0.0f : __expf(p);
  }
}

// -------- deg[bt*512+j] = column sums of A --------------------------------

__global__ void colsum_kernel(const float* __restrict__ A, float* __restrict__ deg)
{
  int idx = blockIdx.x * blockDim.x + threadIdx.x;
  if (idx >= 16384) return;
  int bt = idx >> 9, j = idx & 511;
  const float* base = A + ((long)bt << 18) + j;
  float s = 0.f;
  for (int i2 = 0; i2 < 512; ++i2) s += base[(long)i2 << 9];
  deg[idx] = s;
}

// -------- Lbar: row0 = f, else -A + diag(deg) ------------------------------

__global__ void lbar_kernel(const float* __restrict__ A, const float* __restrict__ deg,
                            const float* __restrict__ f, float* __restrict__ Lb)
{
  for (long idx = (long)blockIdx.x * blockDim.x + threadIdx.x; idx < 8388608L;
       idx += (long)gridDim.x * blockDim.x) {
    long bt = idx >> 18; long rem = idx & 262143L;
    int i = (int)(rem >> 9), j = (int)(rem & 511);
    float v;
    if (i == 0) v = f[(bt << 9) + j];
    else { v = -A[idx]; if (i == j) v += deg[(bt << 9) + i]; }
    Lb[idx] = v;
  }
}

// -------- Gauss-Jordan inverse, one block (512 thr) per 512x512 matrix -----

__global__ __launch_bounds__(512) void gj_kernel(float* __restrict__ Wm, float* __restrict__ Vm)
{
  const int bt = blockIdx.x, tid = threadIdx.x;
  float* W = Wm + ((long)bt << 18);
  float* V = Vm + ((long)bt << 18);
  __shared__ float pivW[512], pivV[512], fcol[512];
  for (int r = 0; r < 512; ++r) V[((long)r << 9) + tid] = (r == tid) ? 1.0f : 0.0f;
  __syncthreads();
  for (int p = 0; p < 512; ++p) {
    float inv = 1.0f / W[((long)p << 9) + p];
    pivW[tid] = W[((long)p << 9) + tid] * inv;
    pivV[tid] = V[((long)p << 9) + tid] * inv;
    fcol[tid] = W[((long)tid << 9) + p];      // column p snapshot (no races)
    __syncthreads();
    W[((long)p << 9) + tid] = pivW[tid];
    V[((long)p << 9) + tid] = pivV[tid];
    for (int r = 0; r < 512; ++r) {
      if (r == p) continue;
      float fv = fcol[r];
      W[((long)r << 9) + tid] -= fv * pivW[tid];
      V[((long)r << 9) + tid] -= fv * pivV[tid];
    }
    __syncthreads();
  }
}

// -------- adj[i,j] = dx[j,i] = A[j,i]*dg[i]*(i!=0) - A[j,i]*LI[i,j]*(j!=0) --

__global__ void dx_kernel(const float* __restrict__ A, const float* __restrict__ LI,
                          float* __restrict__ adj)
{
  for (long idx = (long)blockIdx.x * blockDim.x + threadIdx.x; idx < 8388608L;
       idx += (long)gridDim.x * blockDim.x) {
    long bt = idx >> 18; long rem = idx & 262143L;
    int i = (int)(rem >> 9), j = (int)(rem & 511);
    long base = bt << 18;
    float aji = A [base + ((long)j << 9) + i];
    float lij = LI[base + ((long)i << 9) + j];
    float dgi = LI[base + ((long)i << 9) + i];
    float v = 0.f;
    if (i != 0) v += aji * dgi;
    if (j != 0) v -= aji * lij;
    adj[base + ((long)i << 9) + j] = v;
  }
}

// -------- entmax15 by bisection (50 iters), in place over rows of 512 ------

__global__ void entmax_kernel(float* __restrict__ adj)
{
  int gw = (blockIdx.x * blockDim.x + threadIdx.x) >> 5;
  int lane = threadIdx.x & 31;
  if (gw >= 16384) return;
  float* row = adj + (long)gw * 512;
  float xs[16]; float mx = -3.0e38f;
#pragma unroll
  for (int c = 0; c < 16; ++c) {
    float v = row[c * 32 + lane] * 0.5f;      // x*(alpha-1)
    xs[c] = v; mx = fmaxf(mx, v);
  }
  mx = wred_max(mx);
  float tau = mx - 1.0f;
  float dm  = 1.0f - 0.04419417382f;          // tau_hi - tau_lo, d = 512
  for (int it = 0; it < 50; ++it) {
    dm *= 0.5f;
    float tm = tau + dm;
    float s = 0.f;
#pragma unroll
    for (int c = 0; c < 16; ++c) { float t = fmaxf(xs[c] - tm, 0.f); s += t * t; }
    s = wred_sum(s);
    if (s - 1.0f >= 0.f) tau = tm;
  }
  float p[16]; float s = 0.f;
#pragma unroll
  for (int c = 0; c < 16; ++c) { float t = fmaxf(xs[c] - tau, 0.f); p[c] = t * t; s += p[c]; }
  s = wred_sum(s);
  float inv = 1.0f / s;
#pragma unroll
  for (int c = 0; c < 16; ++c) row[c * 32 + lane] = p[c] * inv;
}

// -------- denom[t*8192 + b*512 + l] = 1 + row-sum of adj -------------------

__global__ void denom_kernel(const float* __restrict__ adj, float* __restrict__ denom)
{
  int gw = (blockIdx.x * blockDim.x + threadIdx.x) >> 5;
  int lane = threadIdx.x & 31;
  if (gw >= 16384) return;
  int bt = gw >> 9, l = gw & 511, b = bt >> 1, t = bt & 1;
  const float* row = adj + (long)gw * 512;
  float s = 0.f;
  for (int j = lane; j < 512; j += 32) s += row[j];
  s = wred_sum(s);
  if (lane == 0) denom[t * 8192 + b * 512 + l] = s + 1.0f;
}

// -------- out_mask helpers -------------------------------------------------

__global__ void mrow_kernel(const float* __restrict__ adj, float* __restrict__ rowS)
{
  int gw = (blockIdx.x * blockDim.x + threadIdx.x) >> 5;
  int lane = threadIdx.x & 31;
  if (gw >= 8192) return;
  int b = gw >> 9, i = gw & 511;
  const float* r0 = adj + ((long)((b * 2 + 0) * 512 + i)) * 512;
  const float* r1 = adj + ((long)((b * 2 + 1) * 512 + i)) * 512;
  float s = 0.f;
  for (int j = lane; j < 512; j += 32) s += r0[j] + r1[j];
  s = wred_sum(s);
  if (lane == 0) rowS[gw] = s;
}
__global__ void mcol_kernel(const float* __restrict__ adj, float* __restrict__ colS)
{
  int idx = blockIdx.x * blockDim.x + threadIdx.x;
  if (idx >= 8192) return;
  int b = idx >> 9, j = idx & 511;
  float s = 0.f;
  for (int i = 0; i < 512; ++i) {
    s += adj[((long)((b * 2 + 0) * 512 + i)) * 512 + j];
    s += adj[((long)((b * 2 + 1) * 512 + i)) * 512 + j];
  }
  colS[idx] = s;
}
__global__ void mwrite_kernel(const float* __restrict__ rowS, const float* __restrict__ colS,
                              float* __restrict__ out)
{
  int idx = blockIdx.x * blockDim.x + threadIdx.x;
  if (idx >= 8192) return;
  out[idx] = ((rowS[idx] + colS[idx]) == 0.0f) ? 1.0f : 0.0f;
}

// ---------------------------------------------------------------------------
// Host orchestration
// ---------------------------------------------------------------------------

static const int NGLa[2] = {2, 4};
static const int HDa[2]  = {384, 192};

// parameter leaf indices assuming JAX sorted-key pytree flattening:
// [x, mask, agg(W,b), dep0{k(W,b),q(W,b),roots[(W,b)x2]}, dep1{...},
//  gcn0_0{lin(W,b), w[t][l](W,b)}, gcn0_1, gcn1_0, gcn1_1]
static const int DEPK_W[2] = {4, 12}, DEPK_B[2] = {5, 13};
static const int DEPQ_W[2] = {6, 14}, DEPQ_B[2] = {7, 15};
static const int ROOT_W[2][2] = {{8, 10}, {16, 18}};
static const int ROOT_B[2][2] = {{9, 11}, {17, 19}};
static const int GLIN[2][2]   = {{20, 30}, {48, 58}};
static inline int GW_IDX(int layer, int gi, int tree, int l) {
  static const int base[2][2] = {{22, 32}, {50, 60}};
  return base[layer][gi] + (tree * NGLa[gi] + l) * 2;
}

static void launch_gemm(hipStream_t st, const u16* A, const u16* B, float* C,
                        const float* bias, const float* rowdiv,
                        int M, int N, int K, int lda, int ldb, int ldc,
                        long sAo, long sAi, int innerN, long sBo, long sBi,
                        long sCo, long sCi, int batch,
                        float alpha, float bscale, int transB, int relu)
{
  dim3 grid((N + BN - 1) / BN, (M + BM - 1) / BM, batch);
  wmma_gemm_kernel<<<grid, dim3(256), 0, st>>>(A, B, C, bias, rowdiv,
      M, N, K, lda, ldb, ldc, sAo, sAi, innerN, sBo, sBi, sCo, sCi,
      alpha, bscale, transB, relu);
}

static void launch_ew(hipStream_t st, const float* a, int lda, const float* b, int ldb,
                      float* df, int lddf, u16* dh, int lddh,
                      long rows, long cols, int dhFromSum)
{
  long total = rows * cols;
  long blocks = (total + 255) / 256; if (blocks > 8192) blocks = 8192;
  ew_kernel<<<dim3((unsigned)blocks), dim3(256), 0, st>>>(a, lda, b, ldb, df, lddf,
                                                          dh, lddh, rows, cols, dhFromSum);
}

extern "C" void kernel_launch(void* const* d_in, const int* in_sizes, int n_in,
                              void* d_out, int out_size, void* d_ws, size_t ws_size,
                              hipStream_t stream)
{
  (void)in_sizes; (void)n_in; (void)out_size; (void)ws_size;
  const float* x   = (const float*)d_in[0];
  const int* mask  = (const int*)d_in[1];
  auto P = [&](int i) -> const float* { return (const float*)d_in[i]; };

  // ---- workspace layout (~480 MB; lifetime-based aliasing) ----
  char* wsb = (char*)d_ws;
  size_t off = 0;
  auto alloc = [&](size_t bytes) -> char* {
    char* p = wsb + off; off += (bytes + 255) & ~(size_t)255; return p;
  };
  u16*   xbf    = (u16*)  alloc(8192L * 768 * 2);
  u16*   qbf    = (u16*)  alloc(8192L * 768 * 2);
  u16*   kbf    = (u16*)  alloc(8192L * 768 * 2);
  u16*   adjbf  = (u16*)  alloc(16384L * 512 * 2);
  u16*   catbf  = (u16*)  alloc(8192L * 1536 * 2);
  u16*   Sbf    = (u16*)  alloc(8192L * 1536 * 2);
  u16*   headbf = (u16*)  alloc(8192L * 1536 * 2);
  u16*   gabf   = (u16*)  alloc(8192L * 3072 * 2);
  u16*   wslot  = (u16*)  alloc(3072L * 768 * 2);
  float* qf     = (float*)alloc(8192L * 768 * 4);
  float* kf     = (float*)alloc(8192L * 768 * 4);
  float* catf   = qf;                               // alias: spans qf+kf (50.3MB)
  float* scoresA= (float*)alloc(16384L * 512 * 4);
  float* Lbar   = (float*)alloc(16384L * 512 * 4);
  float* LLinv  = (float*)alloc(16384L * 512 * 4);
  float* Axf    = scoresA;                          // alias: scoresA..Lbar (50.3MB)
  float* headf  = (float*)((char*)scoresA + 50331648); // alias: rest of Lbar+LLinv
  float* adjf   = (float*)alloc(16384L * 512 * 4);
  float* gcnall = (float*)alloc(8192L * 3072 * 4);
  float* froots = (float*)alloc(16384L * 4);
  float* deg    = (float*)alloc(16384L * 4);
  float* denom  = (float*)alloc(2L * 8192 * 4);
  float* rowS   = (float*)alloc(8192L * 4);
  float* colS   = (float*)alloc(8192L * 4);

  auto cvt_w = [&](const float* Wsrc, int Krows, int Ncols) {
    launch_ew(stream, Wsrc, Ncols, nullptr, 0, nullptr, 0, wslot, Ncols, Krows, Ncols, 0);
  };

  const long LL = 262144;                 // 512*512
  const dim3 wave_grid(2048), wave_blk(256);
  const dim3 half_grid(1024);

  // x -> bf16 (shared by Q/K projections and root scores use f32 x directly)
  launch_ew(stream, x, 768, nullptr, 0, nullptr, 0, xbf, 768, 8192, 768, 0);

  for (int layer = 0; layer < 2; ++layer) {
    // ---- StructuredAttention ----
    cvt_w(P(DEPQ_W[layer]), 768, 768);
    launch_gemm(stream, xbf, wslot, qf, P(DEPQ_B[layer]), nullptr,
                8192, 768, 768, 768, 768, 768, 0, 0, 1, 0, 0, 0, 0, 1, 1.f, 1.f, 0, 0);
    cvt_w(P(DEPK_W[layer]), 768, 768);
    launch_gemm(stream, xbf, wslot, kf, P(DEPK_B[layer]), nullptr,
                8192, 768, 768, 768, 768, 768, 0, 0, 1, 0, 0, 0, 0, 1, 1.f, 1.f, 0, 0);
    launch_ew(stream, qf, 768, nullptr, 0, nullptr, 0, qbf, 768, 8192, 768, 0);
    launch_ew(stream, kf, 768, nullptr, 0, nullptr, 0, kbf, 768, 8192, 768, 0);
    root_kernel<<<wave_grid, wave_blk, 0, stream>>>(x, P(ROOT_W[layer][0]), P(ROOT_B[layer][0]),
                                                    P(ROOT_W[layer][1]), P(ROOT_B[layer][1]), froots);
    // scores[b,t] = q_t @ k_t^T / sqrt(384)   (z = b*2+t, 32 batches)
    launch_gemm(stream, qbf, kbf, scoresA, nullptr, nullptr,
                512, 512, 384, 768, 768, 512,
                512L * 768, 384, 2, 512L * 768, 384, 2L * LL, LL, 32,
                0.05103103858f, 1.f, /*transB=*/1, 0);
    softmax_expA_kernel<<<wave_grid, wave_blk, 0, stream>>>(scoresA, mask);
    colsum_kernel<<<dim3(64), wave_blk, 0, stream>>>(scoresA, deg);
    lbar_kernel<<<dim3(16384), wave_blk, 0, stream>>>(scoresA, deg, froots, Lbar);
    gj_kernel<<<dim3(32), dim3(512), 0, stream>>>(Lbar, LLinv);
    dx_kernel<<<dim3(16384), wave_blk, 0, stream>>>(scoresA, LLinv, adjf);
    entmax_kernel<<<wave_grid, wave_blk, 0, stream>>>(adjf);
    launch_ew(stream, adjf, 512, nullptr, 0, nullptr, 0, adjbf, 512, 16384, 512, 0);
    denom_kernel<<<wave_grid, wave_blk, 0, stream>>>(adjf, denom);

    // ---- MultiGraphConvolutions ----
    for (int gi = 0; gi < 2; ++gi) {
      const int ngl = NGLa[gi], hd = HDa[gi];
      for (int tree = 0; tree < 2; ++tree) {
        // cache = [x]
        launch_ew(stream, x, 768, nullptr, 0, catf, 1536, catbf, 1536, 8192, 768, 0);
        int catD = 768;
        for (int l = 0; l < ngl; ++l) {
          // Ax[b] = adj_tree[b] @ cat[b]   (16 batches)
          launch_gemm(stream, adjbf + (long)tree * LL, catbf, Axf, nullptr, nullptr,
                      512, catD, 512, 512, 1536, 1536,
                      2L * LL, 0, 1, 512L * 1536, 0, 512L * 1536, 0, 16,
                      1.f, 1.f, 0, 0);
          // Sbf = bf16(Ax + cat)
          launch_ew(stream, Axf, 1536, catf, 1536, nullptr, 0, Sbf, 1536, 8192, catD, 1);
          // g = relu((S@W + 2b)/denom) -> new cache columns
          cvt_w(P(GW_IDX(layer, gi, tree, l)), catD, hd);
          launch_gemm(stream, Sbf, wslot, catf + catD,
                      P(GW_IDX(layer, gi, tree, l) + 1), denom + tree * 8192,
                      8192, hd, catD, 1536, hd, 1536,
                      0, 0, 1, 0, 0, 0, 0, 1, 1.f, 2.f, 0, 1);
          // head slice = g + x slice ; catbf new cols = bf16(g)
          launch_ew(stream, catf + catD, 1536, x + l * hd, 768,
                    headf + tree * 768 + l * hd, 1536, catbf + catD, 1536,
                    8192, hd, 0);
          catD += hd;
        }
      }
      // lin: concat(heads) @ W + b  -> gcn_outputs slot
      launch_ew(stream, headf, 1536, nullptr, 0, nullptr, 0, headbf, 1536, 8192, 1536, 0);
      cvt_w(P(GLIN[layer][gi]), 1536, 768);
      int slot = layer * 2 + gi;
      launch_gemm(stream, headbf, wslot, gcnall + slot * 768,
                  P(GLIN[layer][gi] + 1), nullptr,
                  8192, 768, 1536, 1536, 768, 3072,
                  0, 0, 1, 0, 0, 0, 0, 1, 1.f, 1.f, 0, 0);
    }
  }

  // ---- aggregation: concat(gcn_outputs) @ aggW + aggB -> gcn_output ----
  launch_ew(stream, gcnall, 3072, nullptr, 0, nullptr, 0, gabf, 3072, 8192, 3072, 0);
  cvt_w(P(2), 3072, 768);
  launch_gemm(stream, gabf, wslot, (float*)d_out, P(3), nullptr,
              8192, 768, 3072, 3072, 768, 768,
              0, 0, 1, 0, 0, 0, 0, 1, 1.f, 1.f, 0, 0);

  // ---- out_mask from final layer's (entmax'd) adjacencies ----
  mrow_kernel<<<half_grid, wave_blk, 0, stream>>>(adjf, rowS);
  mcol_kernel<<<dim3(32), wave_blk, 0, stream>>>(adjf, colS);
  mwrite_kernel<<<dim3(32), wave_blk, 0, stream>>>(rowS, colS,
                                                   (float*)d_out + 8192L * 768);
}
